// WanSelfAttention_36721970381063
// MI455X (gfx1250) — matrix operand order repaired and checked by reading, not compiled, not run on hardware
//
#include <hip/hip_runtime.h>

// ---------------------------------------------------------------------------
// WanSelfAttention for MI455X (gfx1250): bf16 WMMA GEMMs (TDM/LDS-tiled) +
// flash attention.  S=4096, DIM=3072, N=24 heads, D=128, C=64 rope pairs.
// ---------------------------------------------------------------------------

#define S_LEN 4096
#define DIM   3072
#define NHEAD 24
#define HD    128
#define HC    64

typedef __bf16 bf16;
typedef __attribute__((ext_vector_type(8)))  __bf16 v8bf;
typedef __attribute__((ext_vector_type(16))) __bf16 v16bf;
typedef __attribute__((ext_vector_type(8)))  float  v8f;
typedef __attribute__((ext_vector_type(4)))  unsigned int v4u;
typedef __attribute__((ext_vector_type(8)))  int v8i;
typedef __attribute__((ext_vector_type(4)))  int v4i;

#if __has_builtin(__builtin_amdgcn_tensor_load_to_lds)
#define HAVE_TDM 1
#else
#define HAVE_TDM 0
#endif

union BF16x16 {
    v16bf v;
    struct { v8bf lo; v8bf hi; } h;
};

// A-fragment (16x32 bf16, M x K): lane m=lane%16 holds row m; hi=lane/16
// selects K half. elements 0..15 <- K = k0 + hi*16 + e.
__device__ __forceinline__ v16bf load_a_frag(const bf16* rowptr, int k0, int hi) {
    BF16x16 u;
    u.h.lo = *(const v8bf*)(rowptr + k0 + hi * 16);
    u.h.hi = *(const v8bf*)(rowptr + k0 + hi * 16 + 8);
    return u.v;
}

__device__ __forceinline__ v8f wmma_bf16(v16bf a, v16bf b, v8f c) {
    return __builtin_amdgcn_wmma_f32_16x16x32_bf16(
        /*neg_a=*/false, a, /*neg_b=*/false, b,
        /*c_mod=*/(short)0, c, /*reuse_a=*/false, /*reuse_b=*/false);
}

#if HAVE_TDM
// Issue a TDM 2-D tile load: tile_d1 rows x tile_d0 elements (bf16) from a
// row-major tensor with row stride `stride0` (elements) into LDS (compacted,
// row stride = tile_d0).  Descriptor packed per CDNA5 ISA section 8.3/8.4.
__device__ __forceinline__ void tdm_load_2d(unsigned lds_addr, const bf16* gaddr,
                                            unsigned tensor_d0, unsigned tensor_d1,
                                            unsigned tile_d0, unsigned tile_d1,
                                            unsigned long long stride0) {
    unsigned long long ga = (unsigned long long)(uintptr_t)gaddr;
    v4u g0;
    g0[0] = 1u;                                  // count=1, user mode, no gather
    g0[1] = lds_addr;                            // lds_addr [63:32]
    g0[2] = (unsigned)(ga & 0xffffffffu);        // global_addr [95:64]
    g0[3] = (unsigned)((ga >> 32) & 0x01ffffffu) // global_addr [120:96]
          | (2u << 30);                          // type=2 ("image") [127:126]
    v8i g1;
    g1[0] = (int)(1u << 16);                     // data_size=1 (2 bytes)
    g1[1] = (int)((tensor_d0 & 0xffffu) << 16);  // tensor_dim0 [79:48] lo
    g1[2] = (int)((tensor_d0 >> 16) | ((tensor_d1 & 0xffffu) << 16));
    g1[3] = (int)((tensor_d1 >> 16) | (tile_d0 << 16));   // tile_dim0 [127:112]
    g1[4] = (int)(tile_d1 & 0xffffu);            // tile_dim1 [143:128]
    g1[5] = (int)(stride0 & 0xffffffffu);        // tensor_dim0_stride [207:160]
    g1[6] = (int)((stride0 >> 32) & 0xffffu);
    g1[7] = 0;
    v4i z4 = {0, 0, 0, 0};
#if __clang_major__ >= 23
    v8i z8 = {0, 0, 0, 0, 0, 0, 0, 0};
    __builtin_amdgcn_tensor_load_to_lds(g0, g1, z4, z4, z8, 0);
#else
    __builtin_amdgcn_tensor_load_to_lds(g0, g1, z4, z4, 0);
#endif
}
#endif

// ---------------------------------------------------------------------------
// 1) fp32 -> bf16 elementwise convert
// ---------------------------------------------------------------------------
__global__ void cvt_f32_bf16(const float* __restrict__ src,
                             bf16* __restrict__ dst, int n) {
    int i = blockIdx.x * blockDim.x + threadIdx.x;
    if (i < n) dst[i] = (bf16)src[i];
}

// ---------------------------------------------------------------------------
// 2) GEMM: C[m][o] = sum_k A[m][k] * W[o][k] + bias[o]
//    A: M x K bf16 row-major, W: Nn x K bf16 row-major, C: M x Nn fp32.
//    Block = 4 waves; block tile 128x128; K step 64, staged through LDS.
//    TDM double-buffered async tile loads when available; cooperative
//    global->LDS staging otherwise.
// ---------------------------------------------------------------------------
__global__ void __launch_bounds__(128)
gemm_bf16_wmma(const bf16* __restrict__ A, const bf16* __restrict__ W,
               const float* __restrict__ bias, float* __restrict__ C,
               int M, int Nn, int K) {
    const int tid  = threadIdx.x;
    const int wave = tid >> 5;
    const int lane = tid & 31;
    const int m    = lane & 15;
    const int hi   = lane >> 4;

    const int row0 = blockIdx.y * 128;
    const int col0 = blockIdx.x * 128;
    const int wr   = (wave >> 1) * 64;   // wave row offset in block tile
    const int wc   = (wave & 1) * 64;    // wave col offset in block tile

    __shared__ __align__(16) bf16 As[2][128 * 64];
    __shared__ __align__(16) bf16 Bs[2][128 * 64];

    v8f acc[4][4];
#pragma unroll
    for (int mi = 0; mi < 4; ++mi)
#pragma unroll
        for (int ni = 0; ni < 4; ++ni)
#pragma unroll
            for (int r = 0; r < 8; ++r) acc[mi][ni][r] = 0.f;

    const int nkb = K / 64;

#if HAVE_TDM
    const unsigned ldsA[2] = {(unsigned)(uintptr_t)&As[0][0],
                              (unsigned)(uintptr_t)&As[1][0]};
    const unsigned ldsB[2] = {(unsigned)(uintptr_t)&Bs[0][0],
                              (unsigned)(uintptr_t)&Bs[1][0]};
    if (wave == 0) {
        tdm_load_2d(ldsA[0], A + (size_t)row0 * K, K, M, 64, 128, (unsigned)K);
        tdm_load_2d(ldsB[0], W + (size_t)col0 * K, K, Nn, 64, 128, (unsigned)K);
    }
#endif

    for (int kb = 0; kb < nkb; ++kb) {
        const int k0 = kb * 64;
#if HAVE_TDM
        const int cur = kb & 1;
        const int nxt = cur ^ 1;
        if (wave == 0) {
            if (kb + 1 < nkb) {
                tdm_load_2d(ldsA[nxt], A + (size_t)row0 * K + k0 + 64, K, M,
                            64, 128, (unsigned)K);
                tdm_load_2d(ldsB[nxt], W + (size_t)col0 * K + k0 + 64, K, Nn,
                            64, 128, (unsigned)K);
#if __has_builtin(__builtin_amdgcn_s_wait_tensorcnt)
                __builtin_amdgcn_s_wait_tensorcnt(2);  // current buffer done
#else
                asm volatile("s_wait_tensorcnt 0x2" ::: "memory");
#endif
            } else {
#if __has_builtin(__builtin_amdgcn_s_wait_tensorcnt)
                __builtin_amdgcn_s_wait_tensorcnt(0);
#else
                asm volatile("s_wait_tensorcnt 0x0" ::: "memory");
#endif
            }
        }
        __syncthreads();
        const bf16* tileA = &As[cur][0];
        const bf16* tileB = &Bs[cur][0];
#else
        // Cooperative staging: one thread per tile row, 64 contiguous bf16.
        {
            const bf16* gA = A + (size_t)(row0 + tid) * K + k0;
            const bf16* gB = W + (size_t)(col0 + tid) * K + k0;
#pragma unroll
            for (int u = 0; u < 8; ++u) {
                *(v8bf*)&As[0][tid * 64 + u * 8] = *(const v8bf*)(gA + u * 8);
                *(v8bf*)&Bs[0][tid * 64 + u * 8] = *(const v8bf*)(gB + u * 8);
            }
        }
        __syncthreads();
        const bf16* tileA = &As[0][0];
        const bf16* tileB = &Bs[0][0];
#endif

#pragma unroll
        for (int kk = 0; kk < 2; ++kk) {   // two K=32 WMMA steps per tile
            v16bf a[4];
#pragma unroll
            for (int mi = 0; mi < 4; ++mi)
                a[mi] = load_a_frag(tileA + (wr + mi * 16 + m) * 64, kk * 32, hi);
#pragma unroll
            for (int ni = 0; ni < 4; ++ni) {
                v16bf b = *(const v16bf*)(tileB + (wc + ni * 16 + m) * 64 +
                                          kk * 32 + hi * 16);
#pragma unroll
                for (int mi = 0; mi < 4; ++mi)
                    acc[mi][ni] = wmma_bf16(a[mi], b, acc[mi][ni]);
            }
        }
        __syncthreads();   // tile consumed before it is overwritten
    }

#pragma unroll
    for (int ni = 0; ni < 4; ++ni) {
        const int col = col0 + wc + ni * 16 + m;
        const float bv = bias[col];
#pragma unroll
        for (int mi = 0; mi < 4; ++mi) {
#pragma unroll
            for (int r = 0; r < 8; ++r) {
                const int row = row0 + wr + mi * 16 + hi * 8 + r;
                C[(size_t)row * Nn + col] = acc[mi][ni][r] + bv;
            }
        }
    }
}

// ---------------------------------------------------------------------------
// 3) RMS-norm over DIM + RoPE, output head-major bf16 [N][S][D]
// ---------------------------------------------------------------------------
__global__ void __launch_bounds__(256)
norm_rope_kernel(const float* __restrict__ src,        // [S][DIM]
                 const float* __restrict__ nw,         // [DIM]
                 const float* __restrict__ fcos,       // [S][C]
                 const float* __restrict__ fsin,       // [S][C]
                 bf16* __restrict__ dst) {             // [N][S][D]
    const int s = blockIdx.x;
    const float* row = src + (size_t)s * DIM;

    __shared__ float red[256];
    float local = 0.f;
    for (int i = threadIdx.x; i < DIM; i += 256) {
        float v = row[i];
        local += v * v;
    }
    red[threadIdx.x] = local;
    __syncthreads();
    for (int off = 128; off > 0; off >>= 1) {
        if (threadIdx.x < off) red[threadIdx.x] += red[threadIdx.x + off];
        __syncthreads();
    }
    const float rms = rsqrtf(red[0] * (1.0f / DIM) + 1e-6f);

    for (int p = threadIdx.x; p < DIM / 2; p += 256) {
        const int n   = p / HC;
        const int c   = p % HC;
        const int col = p * 2;
        const float re = row[col]     * rms * nw[col];
        const float im = row[col + 1] * rms * nw[col + 1];
        const float ct = fcos[(size_t)s * HC + c];
        const float st = fsin[(size_t)s * HC + c];
        const float ore = re * ct - im * st;
        const float oim = re * st + im * ct;
        bf16* o = dst + ((size_t)n * S_LEN + s) * HD + c * 2;
        o[0] = (bf16)ore;
        o[1] = (bf16)oim;
    }
}

// ---------------------------------------------------------------------------
// 4) V transpose: fp32 [S][DIM] -> bf16 [N][D][S]
// ---------------------------------------------------------------------------
__global__ void v_transpose(const float* __restrict__ src,
                            bf16* __restrict__ dst) {
    size_t i = (size_t)blockIdx.x * blockDim.x + threadIdx.x;
    if (i >= (size_t)S_LEN * DIM) return;
    const int s   = (int)(i / DIM);
    const int col = (int)(i % DIM);
    const int n   = col / HD;
    const int d   = col % HD;
    dst[((size_t)n * HD + d) * S_LEN + s] = (bf16)src[i];
}

// ---------------------------------------------------------------------------
// 5) Flash attention. Grid = (S/64, N). Block = 128 threads (4 waves);
//    each wave owns 16 query rows. Key blocks of 64, online softmax.
//    P staged via LDS to re-layout C-frag -> A-frag. Output bf16 [S][DIM].
// ---------------------------------------------------------------------------
__global__ void __launch_bounds__(128)
flash_attn_kernel(const bf16* __restrict__ qh,   // [N][S][D]
                  const bf16* __restrict__ kh,   // [N][S][D]
                  const bf16* __restrict__ vT,   // [N][D][S]
                  const int*  __restrict__ seq_lens,
                  bf16* __restrict__ attn) {     // [S][DIM]
    const int n     = blockIdx.y;
    const int qblk  = blockIdx.x * 64;
    const int tid   = threadIdx.x;
    const int wave  = tid >> 5;
    const int lane  = tid & 31;
    const int m     = lane & 15;
    const int hi    = lane >> 4;
    const int qbase = qblk + wave * 16;

    __shared__ __align__(16) bf16 Pbuf[4 * 16 * 64];
    bf16* P = Pbuf + wave * 16 * 64;

    const bf16* qb = qh + (size_t)n * S_LEN * HD;
    const bf16* kb = kh + (size_t)n * S_LEN * HD;
    const bf16* vb = vT + (size_t)n * HD * S_LEN;

    v16bf qa[4];
    const bf16* qrow = qb + (size_t)(qbase + m) * HD;
#pragma unroll
    for (int kc = 0; kc < 4; ++kc) qa[kc] = load_a_frag(qrow, kc * 32, hi);

    v8f acc[8];
    float m_row[8], l_row[8];
#pragma unroll
    for (int nd = 0; nd < 8; ++nd)
#pragma unroll
        for (int r = 0; r < 8; ++r) acc[nd][r] = 0.f;
#pragma unroll
    for (int r = 0; r < 8; ++r) { m_row[r] = -3.0e38f; l_row[r] = 0.f; }

    const int L = seq_lens[0];
    const int nblk = (L + 63) / 64;
    const float scale = 0.08838834764831845f; // 1/sqrt(128)

    for (int tb = 0; tb < nblk; ++tb) {
        const int t0 = tb * 64;

        v8f sc[4];
#pragma unroll
        for (int j = 0; j < 4; ++j) {
            v8f s;
#pragma unroll
            for (int r = 0; r < 8; ++r) s[r] = 0.f;
            const bf16* kcol = kb + (size_t)(t0 + j * 16 + m) * HD;
#pragma unroll
            for (int kc = 0; kc < 4; ++kc) {
                v16bf b = *(const v16bf*)(kcol + kc * 32 + hi * 16);
                s = wmma_bf16(qa[kc], b, s);
            }
            sc[j] = s;
        }

        float mx[8];
#pragma unroll
        for (int r = 0; r < 8; ++r) mx[r] = -3.0e38f;
#pragma unroll
        for (int j = 0; j < 4; ++j) {
            const int tcol = t0 + j * 16 + m;
            const bool valid = tcol < L;
#pragma unroll
            for (int r = 0; r < 8; ++r) {
                float v = sc[j][r] * scale;
                if (!valid) v = -3.0e38f;
                sc[j][r] = v;
                mx[r] = fmaxf(mx[r], v);
            }
        }
#pragma unroll
        for (int r = 0; r < 8; ++r) {
            float v = mx[r];
#pragma unroll
            for (int off = 8; off > 0; off >>= 1)
                v = fmaxf(v, __shfl_xor(v, off, 16));
            mx[r] = v;
        }

        float alpha[8], rs[8];
#pragma unroll
        for (int r = 0; r < 8; ++r) {
            const float mn = fmaxf(m_row[r], mx[r]);
            alpha[r] = __expf(m_row[r] - mn);
            m_row[r] = mn;
            rs[r] = 0.f;
        }

        __syncthreads(); // protect previous iteration's P reads
#pragma unroll
        for (int j = 0; j < 4; ++j) {
#pragma unroll
            for (int r = 0; r < 8; ++r) {
                const float p = __expf(sc[j][r] - m_row[r]);
                rs[r] += p;
                P[(hi * 8 + r) * 64 + j * 16 + m] = (bf16)p;
            }
        }
#pragma unroll
        for (int r = 0; r < 8; ++r) {
            float v = rs[r];
#pragma unroll
            for (int off = 8; off > 0; off >>= 1)
                v += __shfl_xor(v, off, 16);
            l_row[r] = l_row[r] * alpha[r] + v;
        }

#pragma unroll
        for (int nd = 0; nd < 8; ++nd)
#pragma unroll
            for (int r = 0; r < 8; ++r) acc[nd][r] *= alpha[r];

        __syncthreads(); // P writes visible before A-frag loads

#pragma unroll
        for (int kc2 = 0; kc2 < 2; ++kc2) {
            BF16x16 pu;
            pu.h.lo = *(const v8bf*)(&P[m * 64 + kc2 * 32 + hi * 16]);
            pu.h.hi = *(const v8bf*)(&P[m * 64 + kc2 * 32 + hi * 16 + 8]);
            const v16bf pa = pu.v;
#pragma unroll
            for (int nd = 0; nd < 8; ++nd) {
                const bf16* vcol =
                    vb + (size_t)(nd * 16 + m) * S_LEN + t0 + kc2 * 32 + hi * 16;
                v16bf vv = *(const v16bf*)vcol;
                acc[nd] = wmma_bf16(pa, vv, acc[nd]);
            }
        }
    }

#pragma unroll
    for (int nd = 0; nd < 8; ++nd) {
        const int dcol = nd * 16 + m;
#pragma unroll
        for (int r = 0; r < 8; ++r) {
            const int srow = qbase + hi * 8 + r;
            const float o = acc[nd][r] / l_row[r];
            attn[(size_t)srow * DIM + n * HD + dcol] = (bf16)o;
        }
    }
}

// ---------------------------------------------------------------------------
// Launcher
// ---------------------------------------------------------------------------
extern "C" void kernel_launch(void* const* d_in, const int* in_sizes, int n_in,
                              void* d_out, int out_size, void* d_ws,
                              size_t ws_size, hipStream_t stream) {
    (void)in_sizes; (void)n_in; (void)out_size; (void)ws_size;

    const float* x   = (const float*)d_in[0];
    const float* wq  = (const float*)d_in[1];
    const float* bq  = (const float*)d_in[2];
    const float* wk  = (const float*)d_in[3];
    const float* bk  = (const float*)d_in[4];
    const float* wv  = (const float*)d_in[5];
    const float* bv  = (const float*)d_in[6];
    const float* wo  = (const float*)d_in[7];
    const float* bo  = (const float*)d_in[8];
    const float* nqw = (const float*)d_in[9];
    const float* nkw = (const float*)d_in[10];
    const float* fc  = (const float*)d_in[11];
    const float* fs  = (const float*)d_in[12];
    const int*   sl  = (const int*)d_in[13];
    float* out = (float*)d_out;

    const size_t SZ  = (size_t)S_LEN * DIM;
    const size_t WSZ = (size_t)DIM * DIM;

    char* ws = (char*)d_ws;
    size_t off = 0;
    bf16* xb   = (bf16*)(ws + off); off += SZ * 2;
    bf16* wqb  = (bf16*)(ws + off); off += WSZ * 2;
    bf16* wkb  = (bf16*)(ws + off); off += WSZ * 2;
    bf16* wvb  = (bf16*)(ws + off); off += WSZ * 2;
    bf16* wob  = (bf16*)(ws + off); off += WSZ * 2;
    float* qraw = (float*)(ws + off); off += SZ * 4;
    float* kraw = (float*)(ws + off); off += SZ * 4;
    float* vraw = (float*)(ws + off); off += SZ * 4;
    bf16* qh   = (bf16*)(ws + off); off += SZ * 2;
    bf16* kh   = (bf16*)(ws + off); off += SZ * 2;
    bf16* vT   = (bf16*)(ws + off); off += SZ * 2;
    bf16* attn = (bf16*)(ws + off); off += SZ * 2;

    {
        const int blk = 256;
        cvt_f32_bf16<<<(int)((SZ + blk - 1) / blk), blk, 0, stream>>>(x, xb, (int)SZ);
        cvt_f32_bf16<<<(int)((WSZ + blk - 1) / blk), blk, 0, stream>>>(wq, wqb, (int)WSZ);
        cvt_f32_bf16<<<(int)((WSZ + blk - 1) / blk), blk, 0, stream>>>(wk, wkb, (int)WSZ);
        cvt_f32_bf16<<<(int)((WSZ + blk - 1) / blk), blk, 0, stream>>>(wv, wvb, (int)WSZ);
        cvt_f32_bf16<<<(int)((WSZ + blk - 1) / blk), blk, 0, stream>>>(wo, wob, (int)WSZ);
    }

    {
        dim3 grid(DIM / 128, S_LEN / 128); // (24, 32)
        gemm_bf16_wmma<<<grid, 128, 0, stream>>>(xb, wqb, bq, qraw, S_LEN, DIM, DIM);
        gemm_bf16_wmma<<<grid, 128, 0, stream>>>(xb, wkb, bk, kraw, S_LEN, DIM, DIM);
        gemm_bf16_wmma<<<grid, 128, 0, stream>>>(xb, wvb, bv, vraw, S_LEN, DIM, DIM);
    }

    norm_rope_kernel<<<S_LEN, 256, 0, stream>>>(qraw, nqw, fc, fs, qh);
    norm_rope_kernel<<<S_LEN, 256, 0, stream>>>(kraw, nkw, fc, fs, kh);

    v_transpose<<<(int)((SZ + 255) / 256), 256, 0, stream>>>(vraw, vT);

    {
        dim3 grid(S_LEN / 64, NHEAD); // (64, 24)
        flash_attn_kernel<<<grid, 128, 0, stream>>>(qh, kh, vT, sl, attn);
    }

    {
        dim3 grid(DIM / 128, S_LEN / 128);
        gemm_bf16_wmma<<<grid, 128, 0, stream>>>(attn, wob, bo, out, S_LEN, DIM, DIM);
    }
}
